// EdgeConv_19456201851242
// MI455X (gfx1250) — matrix operand order repaired and checked by reading, not compiled, not run on hardware
//
#include <hip/hip_runtime.h>
#include <hip/hip_bf16.h>

// ---------------------------------------------------------------------------
// EdgeConv for MI455X (gfx1250, wave32, WMMA).
// h[b,n,k,o] = P[b, idx[b,n,k], o] + G[b,n,o]
//   P = X^T * W1^T   (bf16 WMMA, f32 accum)
//   G = X^T * (W2 - W1)^T
// ---------------------------------------------------------------------------

#define BB 8
#define CC 64
#define NN 2048
#define OO 128
#define KK 20

typedef __attribute__((ext_vector_type(16))) __bf16 v16bf;
typedef __attribute__((ext_vector_type(8)))  float  v8f;

union FragAB { v16bf v; uint4 u[2]; };

// ---------------------------------------------------------------- prep_x ---
// XT[b][n][c] = bf16(x[b][c][n]);  xx[b][n] = sum_c x^2
__global__ void prep_x(const float* __restrict__ x, float* __restrict__ xx,
                       __hip_bfloat16* __restrict__ XT) {
    int t = blockIdx.x * 256 + threadIdx.x;      // 0 .. B*N-1
    int b = t >> 11;
    int n = t & (NN - 1);
    const float* xb = x + (size_t)b * CC * NN + n;
    __hip_bfloat16* xt = XT + (size_t)t * CC;
    float s = 0.f;
#pragma unroll
    for (int c = 0; c < CC; ++c) {
        float v = xb[(size_t)c * NN];
        s += v * v;
        xt[c] = __float2bfloat16(v);
    }
    xx[t] = s;
}

// ---------------------------------------------------------------- prep_w ---
// Wb rows 0..127   : bf16(W1[o][c])          (c < 64)
// Wb rows 128..255 : bf16(W2[o][c]-W1[o][c])
__global__ void prep_w(const float* __restrict__ W, __hip_bfloat16* __restrict__ Wb) {
    int o = threadIdx.x;
    if (o < OO) {
        const float* wr = W + (size_t)o * (2 * CC);
        __hip_bfloat16* w1 = Wb + (size_t)o * CC;
        __hip_bfloat16* w2 = Wb + (size_t)(OO + o) * CC;
#pragma unroll
        for (int c = 0; c < CC; ++c) {
            float a = wr[c];
            float d = wr[CC + c] - a;
            w1[c] = __float2bfloat16(a);
            w2[c] = __float2bfloat16(d);
        }
    }
}

// ------------------------------------------------------------------- knn ---
// pd(i,j) = 2*dot(xi,xj) - |xi|^2 - |xj|^2 ; keep top-20 largest per row i.
__global__ void knn_kernel(const float* __restrict__ x, const float* __restrict__ xx,
                           int* __restrict__ idx) {
    __shared__ float tile[64][65];   // [jj][c], padded: conflict-free
    __shared__ float xxs[64];

    int b  = blockIdx.x >> 3;                 // 8 blocks per batch
    int i  = ((blockIdx.x & 7) << 8) + threadIdx.x;
    const float* xb = x + (size_t)b * CC * NN;

    float xi[CC];
#pragma unroll
    for (int c = 0; c < CC; ++c) xi[c] = xb[(size_t)c * NN + i];
    float xxi = xx[b * NN + i];

    float best[KK];
    int   bidx[KK];
#pragma unroll
    for (int k = 0; k < KK; ++k) { best[k] = -3.0e38f; bidx[k] = 0; }

    for (int j0 = 0; j0 < NN; j0 += 64) {
        __syncthreads();
        for (int u = threadIdx.x; u < 64 * CC; u += 256) {
            int c = u >> 6, jj = u & 63;
            tile[jj][c] = xb[(size_t)c * NN + j0 + jj];
        }
        if (threadIdx.x < 64) xxs[threadIdx.x] = xx[b * NN + j0 + threadIdx.x];
        __syncthreads();

        for (int jj = 0; jj < 64; ++jj) {
            float dot = 0.f;
#pragma unroll
            for (int c = 0; c < CC; ++c) dot += xi[c] * tile[jj][c];
            float v = 2.0f * dot - xxi - xxs[jj];
            int   j = j0 + jj;
            if (v > best[KK - 1]) {            // branch-free shift insert
#pragma unroll
                for (int s = KK - 1; s >= 1; --s) {
                    bool shift = (v > best[s - 1]);
                    bool here  = (v > best[s]);
                    float nb = shift ? best[s - 1] : (here ? v : best[s]);
                    int   ni = shift ? bidx[s - 1] : (here ? j : bidx[s]);
                    best[s] = nb; bidx[s] = ni;
                }
                if (v > best[0]) { best[0] = v; bidx[0] = j; }
            }
        }
    }
    int* op = idx + ((size_t)b * NN + i) * KK;
#pragma unroll
    for (int k = 0; k < KK; ++k) op[k] = bidx[k];
}

// --------------------------------------------------------------- gemm_pg ---
// One wave: 16 rows (points) x full O=128, C=64 reduction via 2x bf16 WMMA
// per tile, for both W1 (-> P) and W2-W1 (-> G).
__global__ void gemm_pg(const __hip_bfloat16* __restrict__ XT,
                        const __hip_bfloat16* __restrict__ Wb,
                        float* __restrict__ P, float* __restrict__ G) {
    int wave = threadIdx.x >> 5;
    int lane = threadIdx.x & 31;
    int lo = lane & 15, hi = lane >> 4;
    int b  = blockIdx.x >> 4;                         // 16 blocks per batch
    int m0 = ((blockIdx.x & 15) << 7) + (wave << 4);  // 128 rows/block

    // A fragments: rows m0..m0+15, k-chunks 0..31 and 32..63.
    // Lane<16 holds K 0..15 +c0 ; lane>=16 holds K 16..31 +c0.
    const uint4* xrow =
        reinterpret_cast<const uint4*>(XT + ((size_t)(b * NN) + m0 + lo) * CC);
    FragAB a0, a1;
    a0.u[0] = xrow[hi * 2];     a0.u[1] = xrow[hi * 2 + 1];
    a1.u[0] = xrow[4 + hi * 2]; a1.u[1] = xrow[4 + hi * 2 + 1];

#pragma unroll
    for (int ot = 0; ot < 8; ++ot) {
        int o0 = ot << 4;
        const uint4* w1row =
            reinterpret_cast<const uint4*>(Wb + (size_t)(o0 + lo) * CC);
        const uint4* w2row =
            reinterpret_cast<const uint4*>(Wb + (size_t)(OO + o0 + lo) * CC);
        // B fragments: col n = lo, lane<16 K = c0+0..15, lane>=16 K = c0+16..31
        FragAB b10, b11, b20, b21;
        b10.u[0] = w1row[hi * 2];     b10.u[1] = w1row[hi * 2 + 1];
        b11.u[0] = w1row[4 + hi * 2]; b11.u[1] = w1row[4 + hi * 2 + 1];
        b20.u[0] = w2row[hi * 2];     b20.u[1] = w2row[hi * 2 + 1];
        b21.u[0] = w2row[4 + hi * 2]; b21.u[1] = w2row[4 + hi * 2 + 1];

        v8f accP = {};
        v8f accG = {};
        accP = __builtin_amdgcn_wmma_f32_16x16x32_bf16(false, a0.v, false, b10.v,
                                                       (short)0, accP, false, false);
        accP = __builtin_amdgcn_wmma_f32_16x16x32_bf16(false, a1.v, false, b11.v,
                                                       (short)0, accP, false, false);
        accG = __builtin_amdgcn_wmma_f32_16x16x32_bf16(false, a0.v, false, b20.v,
                                                       (short)0, accG, false, false);
        accG = __builtin_amdgcn_wmma_f32_16x16x32_bf16(false, a1.v, false, b21.v,
                                                       (short)0, accG, false, false);

        // C/D layout: lane lo = col, VGPR r -> row r + hi*8
        size_t base = ((size_t)(b * NN) + m0 + hi * 8) * OO + o0 + lo;
#pragma unroll
        for (int r = 0; r < 8; ++r) {
            P[base + (size_t)r * OO] = accP[r];
            G[base + (size_t)r * OO] = accG[r];
        }
    }
}

// ----------------------------------------------------------------- stats ---
// Deterministic per-block partial sums of h and h^2 per channel.
__global__ void stats_kernel(const float* __restrict__ P, const float* __restrict__ G,
                             const int* __restrict__ idx, float* __restrict__ part) {
    __shared__ float ls[256], lq[256];
    int t = threadIdx.x;
    int o = t & (OO - 1);
    int half = t >> 7;
    int pair0 = blockIdx.x * 64 + half * 32;

    float s = 0.f, q = 0.f;
    for (int p = 0; p < 32; ++p) {
        int bn = pair0 + p;
        int b  = bn >> 11;
        float g = G[(size_t)bn * OO + o];
        const int* id = idx + (size_t)bn * KK;
        for (int k = 0; k < KK; ++k) {
            float h = P[((size_t)(b * NN + id[k])) * OO + o] + g;
            s += h; q += h * h;
        }
    }
    ls[t] = s; lq[t] = q;
    __syncthreads();
    if (t < OO) {
        part[(size_t)blockIdx.x * 256 + t]      = ls[t] + ls[t + 128];
        part[(size_t)blockIdx.x * 256 + OO + t] = lq[t] + lq[t + 128];
    }
}

__global__ void finalize_kernel(const float* __restrict__ gamma,
                                const float* __restrict__ beta,
                                const float* __restrict__ part,
                                float* __restrict__ stats) {
    int o = threadIdx.x;            // 128 threads
    float s = 0.f, q = 0.f;
    for (int blk = 0; blk < 256; ++blk) {
        s += part[(size_t)blk * 256 + o];
        q += part[(size_t)blk * 256 + OO + o];
    }
    const float inv = 1.0f / (float)(BB * NN * KK);
    float mean = s * inv;
    float var  = q * inv - mean * mean;
    float sc   = gamma[o] * rsqrtf(var + 1e-5f);
    stats[o]        = sc;
    stats[OO + o]   = beta[o] - mean * sc;
}

// ------------------------------------------------------------------- out ---
// y = gelu(h*scale+bias); out[b,o,n] = max_k y ; LDS transpose for store.
__global__ void out_kernel(const float* __restrict__ P, const float* __restrict__ G,
                           const int* __restrict__ idx, const float* __restrict__ stats,
                           float* __restrict__ out) {
    __shared__ float tile[16 * OO];
    int b  = blockIdx.x >> 7;
    int n0 = (blockIdx.x & 127) << 4;
    int t  = threadIdx.x;
    int o  = t & (OO - 1);
    int half = t >> 7;
    float scale = stats[o];
    float bias  = stats[OO + o];

    for (int qy = 0; qy < 8; ++qy) {
        int nl = half * 8 + qy;
        int bn = b * NN + n0 + nl;
        float g = G[(size_t)bn * OO + o];
        const int* id = idx + (size_t)bn * KK;
        float m = -3.402823466e38f;
        for (int k = 0; k < KK; ++k) {
            float h = P[((size_t)(b * NN + id[k])) * OO + o] + g;
            float y = h * scale + bias;
            y = 0.5f * y * (1.0f + erff(y * 0.70710678118654752f));
            m = fmaxf(m, y);
        }
        tile[nl * OO + o] = m;
    }
    __syncthreads();
    for (int u = t; u < 16 * OO; u += 256) {
        int oo = u >> 4, nl = u & 15;
        out[((size_t)b * OO + oo) * NN + n0 + nl] = tile[nl * OO + oo];
    }
}

// ---------------------------------------------------------------------------
extern "C" void kernel_launch(void* const* d_in, const int* in_sizes, int n_in,
                              void* d_out, int out_size, void* d_ws, size_t ws_size,
                              hipStream_t stream) {
    (void)in_sizes; (void)n_in; (void)out_size; (void)ws_size;
    const float* x     = (const float*)d_in[0];
    const float* W     = (const float*)d_in[1];
    const float* gamma = (const float*)d_in[2];
    const float* beta  = (const float*)d_in[3];
    float* out = (float*)d_out;

    char* ws = (char*)d_ws;
    // workspace layout (128B-aligned offsets), total ~20.6 MB
    float*          stats = (float*)(ws + 0);                 //   1 KB (scale,bias)
    float*          xx    = (float*)(ws + 4096);              //  64 KB
    int*            idx   = (int*)  (ws + 69632);             // 1.25 MB
    __hip_bfloat16* XT    = (__hip_bfloat16*)(ws + 1380352);  //   2 MB
    __hip_bfloat16* Wb    = (__hip_bfloat16*)(ws + 3477504);  //  32 KB
    float*          P     = (float*)(ws + 3510272);           //   8 MB
    float*          G     = (float*)(ws + 11898880);          //   8 MB
    float*          part  = (float*)(ws + 20287488);          // 256 KB

    prep_x       <<<BB * NN / 256, 256, 0, stream>>>(x, xx, XT);
    prep_w       <<<1, 128, 0, stream>>>(W, Wb);
    knn_kernel   <<<BB * NN / 256, 256, 0, stream>>>(x, xx, idx);
    gemm_pg      <<<BB * NN / 128, 256, 0, stream>>>(XT, Wb, P, G);
    stats_kernel <<<BB * NN / 64, 256, 0, stream>>>(P, G, idx, part);
    finalize_kernel<<<1, 128, 0, stream>>>(gamma, beta, part, stats);
    out_kernel   <<<BB * (NN / 16), 256, 0, stream>>>(P, G, idx, stats, out);
}
